// InteractionLayer_35021163331841
// MI455X (gfx1250) — matrix operand order repaired and checked by reading, not compiled, not run on hardware
//
#include <hip/hip_runtime.h>

typedef __attribute__((ext_vector_type(16))) __bf16 v16bf;
typedef __attribute__((ext_vector_type(8)))  float  v8f;
typedef __attribute__((ext_vector_type(4)))  unsigned int v4u;

static constexpr int kB  = 8;
static constexpr int kLD = 2048;
static constexpr int kLR = 2048;
static constexpr int kDM = 256;
static constexpr int kDF = 256;

union FragU {
    v16bf v;
    unsigned short s[16];
    v4u q[2];
};

__device__ __forceinline__ unsigned short f32_to_bf16(float f) {
    unsigned int u = __float_as_uint(f);
    unsigned int r = 0x7FFFu + ((u >> 16) & 1u);   // round-to-nearest-even
    return (unsigned short)((u + r) >> 16);
}

// ---------------- f32 -> bf16 conversion (vectorized x4) ----------------
__global__ void cvt_f32_bf16(const float* __restrict__ in,
                             unsigned short* __restrict__ out, int n4) {
    int i = blockIdx.x * blockDim.x + threadIdx.x;
    if (i < n4) {
        float4 f = reinterpret_cast<const float4*>(in)[i];
        uint2 o;
        o.x = (unsigned int)f32_to_bf16(f.x) | ((unsigned int)f32_to_bf16(f.y) << 16);
        o.y = (unsigned int)f32_to_bf16(f.z) | ((unsigned int)f32_to_bf16(f.w) << 16);
        reinterpret_cast<uint2*>(out)[i] = o;
    }
}

// ---------------- WMMA fragment loaders (ISA 7.12.2 layouts) ----------------
// A 16x32 bf16, row-major source with contiguous K.
// lanes 0-15: rows M=lane, VGPR0-3 K=0..7, VGPR4-7 K=16..23; lanes 16-31: +8 K.
__device__ __forceinline__ v16bf load_a_frag(const unsigned short* __restrict__ X,
                                             int ld, int row0, int k0, int lane) {
    int laneHi = lane >> 4;
    int row = row0 + (lane & 15);
    const unsigned short* p = X + (size_t)row * ld + k0 + laneHi * 8;
    FragU u;
    u.q[0] = reinterpret_cast<const v4u*>(p)[0];   // K = base+0..7
    u.q[1] = reinterpret_cast<const v4u*>(p)[2];   // K = base+16..23
    return u.v;
}

// B 32x16 bf16 where the SOURCE is row-major [N, K] (multiply by its transpose).
// lanes 0-15: col N=n0+lane, K=0..7 and 16..23; lanes 16-31: +8 K.
__device__ __forceinline__ v16bf load_bT_frag(const unsigned short* __restrict__ S,
                                              int ld, int n0, int k0, int lane) {
    int laneHi = lane >> 4;
    int col = n0 + (lane & 15);
    const unsigned short* p = S + (size_t)col * ld + k0 + laneHi * 8;
    FragU u;
    u.q[0] = reinterpret_cast<const v4u*>(p)[0];
    u.q[1] = reinterpret_cast<const v4u*>(p)[2];
    return u.v;
}

// ---------------- Projection: Out[m,n] = X[m,:] . W[n,:] + bias[n] ----------------
// grid.x = M/16, block = 128 (4 waves, wave w -> cols [64w, 64w+64)).
__global__ __launch_bounds__(128) void proj_gemm(const unsigned short* __restrict__ X,
                                                 const unsigned short* __restrict__ W,
                                                 const float* __restrict__ bias,
                                                 unsigned short* __restrict__ Out) {
    int lane = threadIdx.x & 31, w = threadIdx.x >> 5;
    int row0 = blockIdx.x * 16;
    int n0 = w * 64;
    v8f acc[4] = {};
    // software-pipelined K loop (K = 256, 8 steps, fully unrolled)
    v16bf aC = load_a_frag(X, 256, row0, 0, lane);
    v16bf bC[4];
#pragma unroll
    for (int j = 0; j < 4; ++j) bC[j] = load_bT_frag(W, 256, n0 + 16 * j, 0, lane);
#pragma unroll
    for (int k0 = 0; k0 < 256; k0 += 32) {
        v16bf aN;
        v16bf bN[4];
        if (k0 + 32 < 256) {
            aN = load_a_frag(X, 256, row0, k0 + 32, lane);
#pragma unroll
            for (int j = 0; j < 4; ++j) bN[j] = load_bT_frag(W, 256, n0 + 16 * j, k0 + 32, lane);
        }
        __builtin_amdgcn_sched_barrier(0);   // keep next-step loads above the WMMAs
#pragma unroll
        for (int j = 0; j < 4; ++j)
            acc[j] = __builtin_amdgcn_wmma_f32_16x16x32_bf16(false, aC, false, bC[j],
                                                             (short)0, acc[j], false, false);
        __builtin_amdgcn_sched_barrier(0);   // keep WMMAs above next-step copies
        if (k0 + 32 < 256) {
            aC = aN;
#pragma unroll
            for (int j = 0; j < 4; ++j) bC[j] = bN[j];
        }
    }
    int laneHi = lane >> 4, laneLo = lane & 15;
#pragma unroll
    for (int j = 0; j < 4; ++j) {
        int col = n0 + 16 * j + laneLo;
        float bv = bias[col];
#pragma unroll
        for (int r = 0; r < 8; ++r) {
            int row = row0 + r + 8 * laneHi;
            Out[(size_t)row * 256 + col] = f32_to_bf16(acc[j][r] + bv);
        }
    }
}

// ---------------- V transpose: vb [B*LD, 256] -> vt [B][256][LD] ----------------
// 32x32 LDS tiles, coalesced both directions. grid = (B*LD/32, 256/32), block = 256.
__global__ __launch_bounds__(256) void transpose_v(const unsigned short* __restrict__ in,
                                                   unsigned short* __restrict__ outT) {
    __shared__ unsigned short tile[32][33];
    int tx = threadIdx.x & 31, ty = threadIdx.x >> 5;   // 32 x 8
    int row0 = blockIdx.x * 32;                          // l (global row)
    int col0 = blockIdx.y * 32;                          // f
#pragma unroll
    for (int i = 0; i < 32; i += 8)
        tile[ty + i][tx] = in[(size_t)(row0 + ty + i) * 256 + col0 + tx];
    __syncthreads();
    int b = row0 >> 11;            // 2048 rows per batch, tiles never straddle
    int l0 = row0 & 2047;
    unsigned short* o = outT + ((size_t)b * 256 + col0) * 2048 + l0;
#pragma unroll
    for (int i = 0; i < 32; i += 8)
        o[(size_t)(ty + i) * 2048 + tx] = tile[tx][ty + i];
}

// ---------------- Scores: S[r,l] = (k[r,:] . q[l,:]) / 16 ----------------
// grid = (LR/16, LD/256), block = 128.
__global__ __launch_bounds__(128) void scores_gemm(const unsigned short* __restrict__ Kb,
                                                   const unsigned short* __restrict__ Qb,
                                                   float* __restrict__ S) {
    int lane = threadIdx.x & 31, w = threadIdx.x >> 5;
    int row0 = blockIdx.x * 16;
    int n0 = blockIdx.y * 256 + w * 64;
    v8f acc[4] = {};
    v16bf aC = load_a_frag(Kb, 256, row0, 0, lane);
    v16bf bC[4];
#pragma unroll
    for (int j = 0; j < 4; ++j) bC[j] = load_bT_frag(Qb, 256, n0 + 16 * j, 0, lane);
#pragma unroll
    for (int k0 = 0; k0 < 256; k0 += 32) {
        v16bf aN;
        v16bf bN[4];
        if (k0 + 32 < 256) {
            aN = load_a_frag(Kb, 256, row0, k0 + 32, lane);
#pragma unroll
            for (int j = 0; j < 4; ++j) bN[j] = load_bT_frag(Qb, 256, n0 + 16 * j, k0 + 32, lane);
        }
        __builtin_amdgcn_sched_barrier(0);
#pragma unroll
        for (int j = 0; j < 4; ++j)
            acc[j] = __builtin_amdgcn_wmma_f32_16x16x32_bf16(false, aC, false, bC[j],
                                                             (short)0, acc[j], false, false);
        __builtin_amdgcn_sched_barrier(0);
        if (k0 + 32 < 256) {
            aC = aN;
#pragma unroll
            for (int j = 0; j < 4; ++j) bC[j] = bN[j];
        }
    }
    int laneHi = lane >> 4, laneLo = lane & 15;
#pragma unroll
    for (int j = 0; j < 4; ++j) {
        int col = n0 + 16 * j + laneLo;
#pragma unroll
        for (int r = 0; r < 8; ++r) {
            int row = row0 + r + 8 * laneHi;
            S[(size_t)row * kLD + col] = acc[j][r] * 0.0625f;   // 1/sqrt(256)
        }
    }
}

// ---------------- Row softmax: f32 scores -> bf16 probabilities ----------------
__global__ __launch_bounds__(256) void softmax_row(const float* __restrict__ S,
                                                   unsigned short* __restrict__ P) {
    __shared__ float red[256];
    int t = threadIdx.x;
    const float* s = S + (size_t)blockIdx.x * kLD;
    unsigned short* p = P + (size_t)blockIdx.x * kLD;
    float v[8];
    float mx = -3.402823466e38f;
#pragma unroll
    for (int i = 0; i < 8; ++i) { v[i] = s[t + 256 * i]; mx = fmaxf(mx, v[i]); }
    red[t] = mx; __syncthreads();
    for (int o = 128; o > 0; o >>= 1) {
        if (t < o) red[t] = fmaxf(red[t], red[t + o]);
        __syncthreads();
    }
    mx = red[0]; __syncthreads();
    float sum = 0.f;
#pragma unroll
    for (int i = 0; i < 8; ++i) { v[i] = __expf(v[i] - mx); sum += v[i]; }
    red[t] = sum; __syncthreads();
    for (int o = 128; o > 0; o >>= 1) {
        if (t < o) red[t] += red[t + o];
        __syncthreads();
    }
    float inv = 1.0f / red[0];
#pragma unroll
    for (int i = 0; i < 8; ++i) p[t + 256 * i] = f32_to_bf16(v[i] * inv);
}

// ---------------- PV GEMM fused with residual + LayerNorm ----------------
// rel_add[r,f] = sum_l P[r,l] * Vt[f,l];  x = rel_in + rel_add;  LN over f (256).
// grid = LR/16, block = 128 (4 waves cover the full 256 cols of a 16-row block).
// K loop unrolled x2 with ping-pong fragment buffers (no rotate copies).
__global__ __launch_bounds__(128) void pv_ln(const unsigned short* __restrict__ P,
                                             const unsigned short* __restrict__ Vt,
                                             const float* __restrict__ relin,
                                             const float* __restrict__ lng,
                                             const float* __restrict__ lnb,
                                             float* __restrict__ out) {
    __shared__ float sh1[4][16];
    __shared__ float sh2[4][16];
    int lane = threadIdx.x & 31, w = threadIdx.x >> 5;
    int laneHi = lane >> 4, laneLo = lane & 15;
    int r0 = blockIdx.x * 16;
    int n0 = w * 64;
    v8f acc[4] = {};
    v16bf aA = load_a_frag(P, kLD, r0, 0, lane);
    v16bf bA[4];
#pragma unroll
    for (int j = 0; j < 4; ++j) bA[j] = load_bT_frag(Vt, kLD, n0 + 16 * j, 0, lane);
    v16bf aB;
    v16bf bB[4];
    for (int k0 = 0; k0 < kLD; k0 += 64) {
        // stage 1: prefetch k0+32 into B set, compute with A set (k0)
        aB = load_a_frag(P, kLD, r0, k0 + 32, lane);
#pragma unroll
        for (int j = 0; j < 4; ++j) bB[j] = load_bT_frag(Vt, kLD, n0 + 16 * j, k0 + 32, lane);
        __builtin_amdgcn_sched_barrier(0);
#pragma unroll
        for (int j = 0; j < 4; ++j)
            acc[j] = __builtin_amdgcn_wmma_f32_16x16x32_bf16(false, aA, false, bA[j],
                                                             (short)0, acc[j], false, false);
        __builtin_amdgcn_sched_barrier(0);
        // stage 2: prefetch k0+64 into A set, compute with B set (k0+32)
        if (k0 + 64 < kLD) {
            aA = load_a_frag(P, kLD, r0, k0 + 64, lane);
#pragma unroll
            for (int j = 0; j < 4; ++j) bA[j] = load_bT_frag(Vt, kLD, n0 + 16 * j, k0 + 64, lane);
        }
        __builtin_amdgcn_sched_barrier(0);
#pragma unroll
        for (int j = 0; j < 4; ++j)
            acc[j] = __builtin_amdgcn_wmma_f32_16x16x32_bf16(false, aB, false, bB[j],
                                                             (short)0, acc[j], false, false);
        __builtin_amdgcn_sched_barrier(0);
    }
    // residual add
    float x[4][8];
#pragma unroll
    for (int j = 0; j < 4; ++j) {
        int col = n0 + 16 * j + laneLo;
#pragma unroll
        for (int r = 0; r < 8; ++r) {
            int row = r0 + r + 8 * laneHi;
            x[j][r] = acc[j][r] + relin[(size_t)row * 256 + col];
        }
    }
    // per-row sum & sumsq across the 4 col-subtiles, then across 16-lane halves
    float s1[8], s2[8];
#pragma unroll
    for (int r = 0; r < 8; ++r) {
        float t = 0.f, t2 = 0.f;
#pragma unroll
        for (int j = 0; j < 4; ++j) { float xv = x[j][r]; t += xv; t2 += xv * xv; }
        s1[r] = t; s2[r] = t2;
    }
#pragma unroll
    for (int m = 8; m >= 1; m >>= 1) {
#pragma unroll
        for (int r = 0; r < 8; ++r) {
            s1[r] += __shfl_xor(s1[r], m);
            s2[r] += __shfl_xor(s2[r], m);
        }
    }
    if (laneLo == 0) {
#pragma unroll
        for (int r = 0; r < 8; ++r) {
            sh1[w][r + 8 * laneHi] = s1[r];
            sh2[w][r + 8 * laneHi] = s2[r];
        }
    }
    __syncthreads();
    float mean[8], rstd[8];
#pragma unroll
    for (int r = 0; r < 8; ++r) {
        int row = r + 8 * laneHi;
        float t1 = sh1[0][row] + sh1[1][row] + sh1[2][row] + sh1[3][row];
        float t2 = sh2[0][row] + sh2[1][row] + sh2[2][row] + sh2[3][row];
        float mu = t1 * (1.0f / 256.0f);
        float var = t2 * (1.0f / 256.0f) - mu * mu;
        mean[r] = mu;
        rstd[r] = rsqrtf(var + 1e-5f);
    }
#pragma unroll
    for (int j = 0; j < 4; ++j) {
        int col = n0 + 16 * j + laneLo;
        float g = lng[col], bb = lnb[col];
#pragma unroll
        for (int r = 0; r < 8; ++r) {
            int row = r0 + r + 8 * laneHi;
            out[(size_t)row * 256 + col] = (x[j][r] - mean[r]) * rstd[r] * g + bb;
        }
    }
}

// ---------------- host launch ----------------
extern "C" void kernel_launch(void* const* d_in, const int* in_sizes, int n_in,
                              void* d_out, int out_size, void* d_ws, size_t ws_size,
                              hipStream_t stream) {
    const float* det  = (const float*)d_in[0];
    const float* rel  = (const float*)d_in[1];
    const float* Wdet = (const float*)d_in[2];
    const float* bdet = (const float*)d_in[3];
    const float* Wrel = (const float*)d_in[4];
    const float* brel = (const float*)d_in[5];
    const float* Wval = (const float*)d_in[6];
    const float* bval = (const float*)d_in[7];
    const float* lng  = (const float*)d_in[8];
    const float* lnb  = (const float*)d_in[9];
    float* out = (float*)d_out;

    const size_t NDET = (size_t)kB * kLD * kDM;   // 4,194,304
    const size_t NREL = (size_t)kB * kLR * kDM;
    const size_t NW   = (size_t)kDF * kDM;        // 65,536

    unsigned short* detb = (unsigned short*)d_ws;
    unsigned short* relb = detb + NDET;
    unsigned short* qb   = relb + NREL;
    unsigned short* kb   = qb   + NDET;
    unsigned short* vb   = kb   + NREL;
    unsigned short* vtb  = vb   + NDET;           // V transposed: [B][DF][LD]
    unsigned short* wdb  = vtb  + NDET;
    unsigned short* wrb  = wdb  + NW;
    unsigned short* wvb  = wrb  + NW;
    float* scoresF       = (float*)(wvb + NW);
    unsigned short* prb  = (unsigned short*)(scoresF + (size_t)kLR * kLD);

    // det_in passthrough (output tuple element 0)
    hipMemcpyAsync(out, det, NDET * sizeof(float), hipMemcpyDeviceToDevice, stream);

    // f32 -> bf16
    cvt_f32_bf16<<<dim3((unsigned)(NDET / 4 + 255) / 256), 256, 0, stream>>>(det, detb, (int)(NDET / 4));
    cvt_f32_bf16<<<dim3((unsigned)(NREL / 4 + 255) / 256), 256, 0, stream>>>(rel, relb, (int)(NREL / 4));
    cvt_f32_bf16<<<dim3((unsigned)(NW / 4 + 255) / 256), 256, 0, stream>>>(Wdet, wdb, (int)(NW / 4));
    cvt_f32_bf16<<<dim3((unsigned)(NW / 4 + 255) / 256), 256, 0, stream>>>(Wrel, wrb, (int)(NW / 4));
    cvt_f32_bf16<<<dim3((unsigned)(NW / 4 + 255) / 256), 256, 0, stream>>>(Wval, wvb, (int)(NW / 4));

    // projections: M = B*L = 16384 rows -> 1024 row-tiles
    proj_gemm<<<dim3(1024), 128, 0, stream>>>(detb, wdb, bdet, qb);
    proj_gemm<<<dim3(1024), 128, 0, stream>>>(relb, wrb, brel, kb);
    proj_gemm<<<dim3(1024), 128, 0, stream>>>(detb, wvb, bval, vb);

    // transpose V for contiguous-K fragments in the PV GEMM
    transpose_v<<<dim3((unsigned)(kB * kLD / 32), kDF / 32), 256, 0, stream>>>(vb, vtb);

    // attention + LN per batch (stream-ordered; scores buffer reused)
    for (int b = 0; b < kB; ++b) {
        const unsigned short* kbb  = kb  + (size_t)b * kLR * kDF;
        const unsigned short* qbb  = qb  + (size_t)b * kLD * kDF;
        const unsigned short* vtbb = vtb + (size_t)b * kDF * kLD;
        scores_gemm<<<dim3(kLR / 16, kLD / 256), 128, 0, stream>>>(kbb, qbb, scoresF);
        softmax_row<<<dim3(kLR), 256, 0, stream>>>(scoresF, prb);
        pv_ln<<<dim3(kLR / 16), 128, 0, stream>>>(prb, vtbb,
                                                  rel + (size_t)b * kLR * kDM,
                                                  lng, lnb,
                                                  out + NDET + (size_t)b * kLR * kDM);
    }
}
